// HSMSSD_30331059045050
// MI455X (gfx1250) — compile-verified, matrix operand order checked by reading, not run on hardware
//
#include <hip/hip_runtime.h>
#include <hip/hip_bf16.h>

typedef __attribute__((ext_vector_type(16))) __bf16 v16bf;
typedef __attribute__((ext_vector_type(8)))  __bf16 v8bf;
typedef __attribute__((ext_vector_type(8)))  float  v8f;

#define DEV __device__ __forceinline__

static constexpr int  B_  = 32;
static constexpr int  DM  = 256;   // d_model
static constexpr int  S_  = 64;    // state
static constexpr int  L_  = 4096;  // H*W
static constexpr int  C3  = 192;   // 3*S
static constexpr int  N2  = B_ * S_;  // 2048 (batch-folded N for weight GEMMs)

// ---------------- bf16 helpers ----------------
DEV __bf16 to_bf16(float f) {
  unsigned u = __builtin_bit_cast(unsigned, f);
  unsigned short h = (unsigned short)((u + 0x7FFFu + ((u >> 16) & 1u)) >> 16);
  return __builtin_bit_cast(__bf16, h);
}

// ---------------- WMMA fragment plumbing ----------------
// A fragment (16xK tile slice, 32 bf16 of K): lane half h reads k in
// {8h..8h+7} (elems 0-7) and {16+8h..16+8h+7} (elems 8-15): two b128 loads.
// B side is stored [n][k] so the identical loader works (symmetric layout).
DEV v16bf load_frag(const __bf16* base, int ld, int mn, int kk, int half) {
  const __bf16* p = base + (long)mn * ld + kk + half * 8;
  v8bf lo = *(const v8bf*)(p);
  v8bf hi = *(const v8bf*)(p + 16);
  return __builtin_shufflevector(lo, hi,
      0, 1, 2, 3, 4, 5, 6, 7, 8, 9, 10, 11, 12, 13, 14, 15);
}

DEV v8f wmma_bf16(v16bf a, v16bf b, v8f c) {
  return __builtin_amdgcn_wmma_f32_16x16x32_bf16(
      false, a, false, b, (short)0, c, false, false);
}

// Register-blocked 32x64 output block per wave: 2 (m) x 4 (n) tiles of 16x16.
// Per 32-wide k-step: 2 A-frags + 4 B-frags feed 8 WMMAs (0.75 frag/WMMA).
DEV void wmma_block_2x4(const __bf16* A, int lda, const __bf16* Bt, int ldb,
                        int row0, int col0, int K, int lane, v8f acc[2][4]) {
  int half = lane >> 4;
  int r = row0 + (lane & 15);
  int c = col0 + (lane & 15);
  for (int kk = 0; kk < K; kk += 32) {
    // stream-ahead hint on the B operand (speculative, dropped if invalid)
    __builtin_prefetch((const void*)(Bt + (long)c * ldb + kk + 128), 0, 0);
    v16bf a0 = load_frag(A, lda, r,      kk, half);
    v16bf a1 = load_frag(A, lda, r + 16, kk, half);
#pragma unroll
    for (int j = 0; j < 4; ++j) {
      v16bf b = load_frag(Bt, ldb, c + 16 * j, kk, half);
      acc[0][j] = wmma_bf16(a0, b, acc[0][j]);
      acc[1][j] = wmma_bf16(a1, b, acc[1][j]);
    }
  }
}

// ---------------- K0a: x -> bf16 (natural + transposed) ----------------
__global__ void k_convert_x(const float* __restrict__ x,
                            __bf16* __restrict__ xbf,
                            __bf16* __restrict__ xTbf) {
  long idx = (long)blockIdx.x * 256 + threadIdx.x;   // B*D*L = 33.5M
  int  l   = (int)(idx & (L_ - 1));
  long t   = idx >> 12;                              // /L_
  int  d   = (int)(t & (DM - 1));
  int  b   = (int)(t >> 8);
  float v  = x[idx];
  __bf16 h = to_bf16(v);
  xbf[idx] = h;
  xTbf[((long)b * L_ + l) * DM + d] = h;
}

// ---------------- K0b: weights -> bf16 ----------------
__global__ void k_convert_w(const float* __restrict__ w1,
                            const float* __restrict__ whz,
                            const float* __restrict__ wout,
                            __bf16* __restrict__ w1b,
                            __bf16* __restrict__ whzb,
                            __bf16* __restrict__ woutb) {
  int idx = blockIdx.x * 256 + threadIdx.x;          // 245760
  if (idx < C3 * DM) {
    w1b[idx] = to_bf16(w1[idx]);
  } else if (idx < C3 * DM + 2 * DM * DM) {
    int i = idx - C3 * DM;
    whzb[i] = to_bf16(whz[i]);
  } else {
    int i = idx - C3 * DM - 2 * DM * DM;
    woutb[i] = to_bf16(wout[i]);
  }
}

// ---------------- K1: bcdt = W1(192x256) @ x(b:256x4096) + b1 ----------------
__global__ void k_gemm_bcdt(const __bf16* __restrict__ w1b,
                            const __bf16* __restrict__ xTbf,
                            const float* __restrict__ bias,
                            float* __restrict__ out) {
  int lane = threadIdx.x & 31;
  int tile = blockIdx.x * 8 + (threadIdx.x >> 5);    // 32*6*64 = 12288 supertiles
  int b    = tile / (6 * 64);
  int rem  = tile - b * (6 * 64);
  int m0   = (rem >> 6) * 32;
  int n0   = (rem & 63) * 64;
  const __bf16* Bt = xTbf + (long)b * L_ * DM;       // [n=l][k=d], ld=DM
  v8f acc[2][4] = {};
  wmma_block_2x4(w1b, DM, Bt, DM, m0, n0, DM, lane, acc);
  int half = lane >> 4;
  float* dst = out + (long)b * C3 * L_;
#pragma unroll
  for (int i = 0; i < 2; ++i)
#pragma unroll
    for (int jt = 0; jt < 4; ++jt) {
      int c = n0 + 16 * jt + (lane & 15);
#pragma unroll
      for (int j = 0; j < 8; ++j) {
        int m = m0 + 16 * i + j + 8 * half;
        dst[(long)m * L_ + c] = acc[i][jt][j] + bias[m];
      }
    }
}

// ---------------- K2: depthwise 3x3 over (64,64), pad 1 ----------------
__global__ void k_dwconv(const float* __restrict__ in,
                         const float* __restrict__ wdw,
                         const float* __restrict__ bdw,
                         float* __restrict__ out) {
  long idx = (long)blockIdx.x * 256 + threadIdx.x;   // B*192*L = 25.2M
  int  l   = (int)(idx & (L_ - 1));
  long t   = idx >> 12;
  int  o   = (int)(t % C3);
  int  b   = (int)(t / C3);
  int  h   = l >> 6, w = l & 63;
  const float* src = in + ((long)b * C3 + o) * L_;
  float acc = bdw[o];
#pragma unroll
  for (int dy = -1; dy <= 1; ++dy) {
#pragma unroll
    for (int dx = -1; dx <= 1; ++dx) {
      int hh = h + dy, ww = w + dx;
      if (hh >= 0 && hh < 64 && ww >= 0 && ww < 64)
        acc += src[hh * 64 + ww] * wdw[o * 9 + (dy + 1) * 3 + (dx + 1)];
    }
  }
  out[idx] = acc;
}

// ---------------- K3: softmax stats over L per (b,s) ----------------
__global__ void k_softmax_stats(const float* __restrict__ bcdtc,
                                const float* __restrict__ Ap,
                                float* __restrict__ rowmax,
                                float* __restrict__ rowsum) {
  __shared__ float red[256];
  int r = blockIdx.x;                                // b*64+s, 2048 rows
  int b = r >> 6, s = r & 63;
  const float* p = bcdtc + ((long)b * C3 + 128 + s) * L_;
  float a = Ap[s];
  float m = -1e30f;
  for (int l = threadIdx.x; l < L_; l += 256) m = fmaxf(m, p[l] + a);
  red[threadIdx.x] = m;
  __syncthreads();
  for (int st = 128; st > 0; st >>= 1) {
    if (threadIdx.x < st)
      red[threadIdx.x] = fmaxf(red[threadIdx.x], red[threadIdx.x + st]);
    __syncthreads();
  }
  m = red[0];
  __syncthreads();
  float sum = 0.f;
  for (int l = threadIdx.x; l < L_; l += 256) sum += __expf(p[l] + a - m);
  red[threadIdx.x] = sum;
  __syncthreads();
  for (int st = 128; st > 0; st >>= 1) {
    if (threadIdx.x < st) red[threadIdx.x] += red[threadIdx.x + st];
    __syncthreads();
  }
  if (threadIdx.x == 0) { rowmax[r] = m; rowsum[r] = red[0]; }
}

// ---------------- K4: AB = softmax(dt+A)*Bm (bf16, S-major); Ct (L,S) ----------------
__global__ void k_pointwise_abc(const float* __restrict__ bcdtc,
                                const float* __restrict__ Ap,
                                const float* __restrict__ rowmax,
                                const float* __restrict__ rowsum,
                                __bf16* __restrict__ ABbf,
                                __bf16* __restrict__ Cmt) {
  long idx = (long)blockIdx.x * 256 + threadIdx.x;   // B*S*L = 8.39M
  int  l   = (int)(idx & (L_ - 1));
  long t   = idx >> 12;
  int  s   = (int)(t & 63);
  int  b   = (int)(t >> 6);
  int  r   = b * 64 + s;
  const float* base = bcdtc + (long)b * C3 * L_;
  float dt = base[(long)(128 + s) * L_ + l];
  float aw = __expf(dt + Ap[s] - rowmax[r]) / rowsum[r];
  float bm = base[(long)s * L_ + l];
  ABbf[((long)b * S_ + s) * L_ + l] = to_bf16(aw * bm);
  float cm = base[(long)(64 + s) * L_ + l];
  Cmt[((long)b * L_ + l) * S_ + s] = to_bf16(cm);
}

// ---------------- K5: hT(n=b*S+s, d) = AB(s,:) . x(d,:) ----------------
__global__ void k_gemm_h(const __bf16* __restrict__ ABbf,
                         const __bf16* __restrict__ xbf,
                         __bf16* __restrict__ hT) {
  int lane = threadIdx.x & 31;
  int tile = blockIdx.x * 8 + (threadIdx.x >> 5);    // 32*2*4 = 256 supertiles
  int b    = tile >> 3;
  int rem  = tile & 7;
  int m0   = (rem >> 2) * 32;                        // s
  int n0   = (rem & 3) * 64;                         // d
  const __bf16* A  = ABbf + (long)b * S_ * L_;       // [m=s][k=l]
  const __bf16* Bt = xbf + (long)b * DM * L_;        // [n=d][k=l]
  v8f acc[2][4] = {};
  wmma_block_2x4(A, L_, Bt, L_, m0, n0, L_, lane, acc);
  int half = lane >> 4;
#pragma unroll
  for (int i = 0; i < 2; ++i)
#pragma unroll
    for (int jt = 0; jt < 4; ++jt) {
      int c = n0 + 16 * jt + (lane & 15);
#pragma unroll
      for (int j = 0; j < 8; ++j) {
        int m = m0 + 16 * i + j + 8 * half;
        hT[((long)b * S_ + m) * DM + c] = to_bf16(acc[i][jt][j]);
      }
    }
}

// ---------------- K6: hz(512,2048) = Whz @ h (batch folded in N) ----------------
__global__ void k_gemm_hz(const __bf16* __restrict__ whzb,
                          const __bf16* __restrict__ hT,
                          const float* __restrict__ bias,
                          float* __restrict__ hz) {
  int lane = threadIdx.x & 31;
  int tile = blockIdx.x * 8 + (threadIdx.x >> 5);    // 16*32 = 512 supertiles
  int m0   = (tile >> 5) * 32;
  int n0   = (tile & 31) * 64;
  v8f acc[2][4] = {};
  wmma_block_2x4(whzb, DM, hT, DM, m0, n0, DM, lane, acc);
  int half = lane >> 4;
#pragma unroll
  for (int i = 0; i < 2; ++i)
#pragma unroll
    for (int jt = 0; jt < 4; ++jt) {
      int c = n0 + 16 * jt + (lane & 15);
#pragma unroll
      for (int j = 0; j < 8; ++j) {
        int m = m0 + 16 * i + j + 8 * half;
        hz[(long)m * N2 + c] = acc[i][jt][j] + bias[m];
      }
    }
}

// ---------------- K7: g = h2*silu(z) + h2*Dp, transposed bf16 ----------------
__global__ void k_silu_gate(const float* __restrict__ hz,
                            const float* __restrict__ Dp,
                            __bf16* __restrict__ gT) {
  int idx = blockIdx.x * 256 + threadIdx.x;          // 256*2048
  int n   = idx & (N2 - 1);
  int m   = idx >> 11;
  float h2  = hz[(long)m * N2 + n];
  float z   = hz[(long)(m + DM) * N2 + n];
  float sig = 1.f / (1.f + __expf(-z));
  float g   = h2 * (z * sig) + h2 * Dp[0];
  gT[(long)n * DM + m] = to_bf16(g);
}

// ---------------- K8: hout(256,2048) = Wout @ g; writes f32 out + bf16 copy ----------------
__global__ void k_gemm_hout(const __bf16* __restrict__ woutb,
                            const __bf16* __restrict__ gT,
                            const float* __restrict__ bias,
                            float* __restrict__ hout_out,   // (B,256,64)
                            __bf16* __restrict__ houtbf) {
  int lane = threadIdx.x & 31;
  int tile = blockIdx.x * 8 + (threadIdx.x >> 5);    // 8*32 = 256 supertiles
  int m0   = (tile >> 5) * 32;
  int n0   = (tile & 31) * 64;
  v8f acc[2][4] = {};
  wmma_block_2x4(woutb, DM, gT, DM, m0, n0, DM, lane, acc);
  int half = lane >> 4;
#pragma unroll
  for (int i = 0; i < 2; ++i)
#pragma unroll
    for (int jt = 0; jt < 4; ++jt) {
      int n = n0 + 16 * jt + (lane & 15);
      int b = n >> 6, s = n & 63;
#pragma unroll
      for (int j = 0; j < 8; ++j) {
        int m = m0 + 16 * i + j + 8 * half;
        float v = acc[i][jt][j] + bias[m];
        long  o = ((long)b * DM + m) * S_ + s;
        hout_out[o] = v;
        houtbf[o]   = to_bf16(v);
      }
    }
}

// ---------------- K9: y(b:256x4096) = hout(b:256x64) @ Cm(b:64x4096) ----------------
__global__ void k_gemm_y(const __bf16* __restrict__ houtbf,
                         const __bf16* __restrict__ Cmt,
                         float* __restrict__ y) {
  int lane = threadIdx.x & 31;
  int tile = blockIdx.x * 8 + (threadIdx.x >> 5);    // 32*8*64 = 16384 supertiles
  int b    = tile >> 9;
  int rem  = tile & 511;
  int m0   = (rem >> 6) * 32;                        // d
  int n0   = (rem & 63) * 64;                        // l
  const __bf16* A  = houtbf + (long)b * DM * S_;     // [m=d][k=s], ld=64
  const __bf16* Bt = Cmt + (long)b * L_ * S_;        // [n=l][k=s], ld=64
  v8f acc[2][4] = {};
  wmma_block_2x4(A, S_, Bt, S_, m0, n0, S_, lane, acc);
  int half = lane >> 4;
  float* dst = y + (long)b * DM * L_;
#pragma unroll
  for (int i = 0; i < 2; ++i)
#pragma unroll
    for (int jt = 0; jt < 4; ++jt) {
      int c = n0 + 16 * jt + (lane & 15);
#pragma unroll
      for (int j = 0; j < 8; ++j) {
        int m = m0 + 16 * i + j + 8 * half;
        dst[(long)m * L_ + c] = acc[i][jt][j];
      }
    }
}

// ---------------- launch ----------------
extern "C" void kernel_launch(void* const* d_in, const int* in_sizes, int n_in,
                              void* d_out, int out_size, void* d_ws, size_t ws_size,
                              hipStream_t stream) {
  (void)in_sizes; (void)n_in; (void)out_size; (void)ws_size;
  const float* x      = (const float*)d_in[0];
  const float* w_bcdt = (const float*)d_in[3];
  const float* b_bcdt = (const float*)d_in[4];
  const float* w_dw   = (const float*)d_in[5];
  const float* b_dw   = (const float*)d_in[6];
  const float* w_hz   = (const float*)d_in[7];
  const float* b_hz   = (const float*)d_in[8];
  const float* w_out  = (const float*)d_in[9];
  const float* b_out  = (const float*)d_in[10];
  const float* A_par  = (const float*)d_in[11];
  const float* D_par  = (const float*)d_in[12];

  float* y_out    = (float*)d_out;                        // (B,256,4096)
  float* hout_out = y_out + (size_t)B_ * DM * L_;         // (B,256,64)

  // workspace carve-up (256B aligned regions)
  char*  w   = (char*)d_ws;
  size_t off = 0;
  auto alloc = [&](size_t bytes) -> void* {
    void* p = w + off;
    off += (bytes + 255) & ~(size_t)255;
    return p;
  };
  __bf16* xbf    = (__bf16*)alloc((size_t)B_ * DM * L_ * 2);   // 67 MB
  __bf16* xTbf   = (__bf16*)alloc((size_t)B_ * L_ * DM * 2);   // 67 MB
  __bf16* w1b    = (__bf16*)alloc((size_t)C3 * DM * 2);
  __bf16* whzb   = (__bf16*)alloc((size_t)2 * DM * DM * 2);
  __bf16* woutb  = (__bf16*)alloc((size_t)DM * DM * 2);
  float*  bcdt   = (float*)alloc((size_t)B_ * C3 * L_ * 4);    // 100 MB
  float*  bcdtc  = (float*)alloc((size_t)B_ * C3 * L_ * 4);    // 100 MB
  float*  rowmax = (float*)alloc((size_t)N2 * 4);
  float*  rowsum = (float*)alloc((size_t)N2 * 4);
  __bf16* ABbf   = (__bf16*)alloc((size_t)B_ * S_ * L_ * 2);   // 16.8 MB
  __bf16* Cmt    = (__bf16*)alloc((size_t)B_ * L_ * S_ * 2);   // 16.8 MB
  __bf16* hT     = (__bf16*)alloc((size_t)N2 * DM * 2);
  float*  hz     = (float*)alloc((size_t)2 * DM * N2 * 4);
  __bf16* gT     = (__bf16*)alloc((size_t)N2 * DM * 2);
  __bf16* houtbf = (__bf16*)alloc((size_t)B_ * DM * S_ * 2);

  dim3 blk(256);

  k_convert_x<<<dim3((B_ * DM * L_) / 256), blk, 0, stream>>>(x, xbf, xTbf);
  k_convert_w<<<dim3((C3 * DM + 3 * DM * DM) / 256), blk, 0, stream>>>(
      w_bcdt, w_hz, w_out, w1b, whzb, woutb);

  k_gemm_bcdt<<<dim3((B_ * 6 * 64) / 8), blk, 0, stream>>>(w1b, xTbf, b_bcdt, bcdt);
  k_dwconv<<<dim3((B_ * C3 * L_) / 256), blk, 0, stream>>>(bcdt, w_dw, b_dw, bcdtc);
  k_softmax_stats<<<dim3(N2), blk, 0, stream>>>(bcdtc, A_par, rowmax, rowsum);
  k_pointwise_abc<<<dim3((B_ * S_ * L_) / 256), blk, 0, stream>>>(
      bcdtc, A_par, rowmax, rowsum, ABbf, Cmt);

  k_gemm_h<<<dim3((B_ * 2 * 4) / 8), blk, 0, stream>>>(ABbf, xbf, hT);
  k_gemm_hz<<<dim3((16 * 32) / 8), blk, 0, stream>>>(whzb, hT, b_hz, hz);
  k_silu_gate<<<dim3((DM * N2) / 256), blk, 0, stream>>>(hz, D_par, gT);
  k_gemm_hout<<<dim3((8 * 32) / 8), blk, 0, stream>>>(woutb, gT, b_out, hout_out, houtbf);
  k_gemm_y<<<dim3((B_ * 8 * 64) / 8), blk, 0, stream>>>(houtbf, Cmt, y_out);
}
